// VRNN_68968584839787
// MI455X (gfx1250) — matrix operand order repaired and checked by reading, not compiled, not run on hardware
//
#include <hip/hip_runtime.h>
#include <cstdint>
#include <cstddef>

typedef __attribute__((ext_vector_type(16))) _Float16 v16h;
typedef __attribute__((ext_vector_type(8)))  float    v8f;

#define XD 2
#define HD 256
#define ZD 64
#define BB 2048
#define NSTEP 63
#define WAVES 4
#define TPB (WAVES * 32)
#define LOG2PI 1.83787706640934548356f

// ---- packed f16 weight offsets in d_ws (units: halves). Block = 512 halves =
// one (kSlice=32, nTile=16) WMMA B tile laid out [lane][half s]: n = lane%16, k = 8*(lane/16)+s+8*(s/8).
constexpr size_t OFF_PHIX1 = 0;                          // N=256 K=2   nT=16 kS=1
constexpr size_t OFF_PHIX2 = OFF_PHIX1 + 16 * 1 * 512;   // N=256 K=256 nT=16 kS=8
constexpr size_t OFF_PHIZ  = OFF_PHIX2 + 16 * 8 * 512;   // N=256 K=64  nT=16 kS=2
constexpr size_t OFF_ENC1  = OFF_PHIZ  + 16 * 2 * 512;   // N=256 K=512 nT=16 kS=16
constexpr size_t OFF_ENC2  = OFF_ENC1  + 16 * 16 * 512;
constexpr size_t OFF_ENCM  = OFF_ENC2  + 16 * 8 * 512;   // N=64 K=256 nT=4 kS=8
constexpr size_t OFF_ENCL  = OFF_ENCM  + 4 * 8 * 512;
constexpr size_t OFF_PR    = OFF_ENCL  + 4 * 8 * 512;
constexpr size_t OFF_PRM   = OFF_PR    + 16 * 8 * 512;
constexpr size_t OFF_PRL   = OFF_PRM   + 4 * 8 * 512;
constexpr size_t OFF_DEC1  = OFF_PRL   + 4 * 8 * 512;
constexpr size_t OFF_DEC2  = OFF_DEC1  + 16 * 16 * 512;
constexpr size_t OFF_DECM  = OFF_DEC2  + 16 * 8 * 512;   // N=2(pad16) K=256 nT=1 kS=8
constexpr size_t OFF_DECL  = OFF_DECM  + 1 * 8 * 512;
constexpr size_t OFF_GIH   = OFF_DECL  + 1 * 8 * 512;    // N=768 K=512 nT=48 kS=16
constexpr size_t OFF_GHH   = OFF_GIH   + 48 * 16 * 512;  // N=768 K=256 nT=48 kS=8
constexpr size_t W_TOTAL   = OFF_GHH   + 48 * 8 * 512;   // 1,212,416 halves (2.4 MB)

// ---------------------------------------------------------------------------
// Prep: pack fp32 weight W (N_out x K_in, row-major, y = x @ W^T) into the
// CDNA5 16x16x32 f16 B-operand layout: within a block, n = lane%16 (0..15),
// half j -> k = 8*(lane/16)+j+8*(j/8).  Out-of-range (padded) entries are zero.
// ---------------------------------------------------------------------------
__global__ void vrnn_pack_w(const float* __restrict__ src, _Float16* __restrict__ dst,
                            int N, int K, int nTiles, int kSlices) {
  const int total = kSlices * nTiles * 512;
  for (int idx = blockIdx.x * blockDim.x + threadIdx.x; idx < total;
       idx += gridDim.x * blockDim.x) {
    const int within = idx & 511;
    const int blk    = idx >> 9;
    const int nt     = blk % nTiles;
    const int ks     = blk / nTiles;
    const int k      = ks * 32 + (((within >> 4) >> 4) << 3) + (((within & 15) >> 3) << 4) + (within & 7);
    const int n      = nt * 16 + ((within >> 4) & 15);
    const float v = (n < N && k < K) ? src[(size_t)n * K + k] : 0.0f;
    dst[idx] = (_Float16)v;
  }
}

__global__ void vrnn_init_out(float* o) {
  if (threadIdx.x < 2) o[threadIdx.x] = 0.0f;
}

// ---------------------------------------------------------------------------
// A-operand loader: 16-bit A 16x32 ISA layout. Row-major f16 LDS tile
// (stride halves), rows 0..15, K slice at kb.  lane<16 -> K {0..7,16..23},
// lane>=16 -> K {8..15,24..31}; VGPR v holds pair k=(v/4)*16+(lane/16)*8+(v%4)*2.
// 8 aligned ds_load_b32 per operand.
// ---------------------------------------------------------------------------
__device__ __forceinline__ v16h load_A(const _Float16* base, int stride, int kb, int lane) {
  const int row = lane & 15, hh = lane >> 4;
  union { v16h h; uint32_t u[8]; } a;
  const uint32_t* p = (const uint32_t*)(base + row * stride + kb);
#pragma unroll
  for (int v = 0; v < 8; ++v)
    a.u[v] = p[((v >> 2) << 3) + (hh << 2) + (v & 3)];
  return a.h;
}

__device__ __forceinline__ v8f wmma16(v16h a, v16h b, v8f c) {
  return __builtin_amdgcn_wmma_f32_16x16x32_f16(false, a, false, b, (short)0, c,
                                                false, false);
}

// Accumulate nSlices K-slices of (A_lds x W_packed) into a 16x16 f32 tile.
__device__ __forceinline__ v8f gemm_part(v8f acc, const _Float16* A, int astride,
                                         const _Float16* W, int nTiles, int nt,
                                         int nSlices, int wSliceOff, int lane) {
#pragma unroll
  for (int s = 0; s < nSlices; ++s) {
    v16h a = load_A(A, astride, s * 32, lane);
    const v16h b = *(const v16h*)(W + ((size_t)((wSliceOff + s) * nTiles + nt) * 32 + lane) * 16);
    acc = wmma16(a, b, acc);
  }
  return acc;
}

// D layout: lane -> n = lane&15, VGPR r -> m = r + 8*(lane>>4).
__device__ __forceinline__ void store_act(_Float16* dst, int stride, int nb, v8f d,
                                          const float* bias, bool lrelu, int lane) {
  const int col = lane & 15, hh = lane >> 4;
  const float b = bias[nb + col];
#pragma unroll
  for (int r = 0; r < 8; ++r) {
    float v = d[r] + b;
    if (lrelu) v = v > 0.0f ? v : 0.01f * v;
    dst[(r + 8 * hh) * stride + nb + col] = (_Float16)v;
  }
}

struct VrnnArgs {
  const float* x;
  const float* eps;
  const float *b_px1, *b_px2, *b_pz, *b_e1, *b_e2, *b_em, *b_el;
  const float *b_pr, *b_pm, *b_pl, *b_d1, *b_d2, *b_dm, *b_dl;
  const _Float16* w;
  float* out;
};

// ---------------------------------------------------------------------------
// Persistent VRNN: each 4-wave workgroup owns 16 batch rows for all 63 steps.
// Layer N-tiles are split across the 4 waves (nt = wave, wave+4, ...), so the
// same 60KB of LDS-resident activations feeds 4x the waves per SIMD for
// latency hiding.  Weights stream from the L2-resident packed f16 pool.
// ---------------------------------------------------------------------------
__global__ __launch_bounds__(TPB) void vrnn_main(VrnnArgs P) {
  __shared__ alignas(16) _Float16 hA[16 * HD], hB[16 * HD];
  __shared__ alignas(16) _Float16 pxb[16 * HD], pzb[16 * HD];
  __shared__ alignas(16) _Float16 t1b[16 * HD], t2b[16 * HD];
  __shared__ alignas(16) _Float16 zbuf[16 * ZD], xbuf[16 * 32];
  __shared__ alignas(16) float emb[16 * ZD], elb[16 * ZD];

  const int tid   = threadIdx.x;
  const int lane  = tid & 31;
  const int wave  = tid >> 5;
  const int mBase = blockIdx.x * 16;
  const int col = lane & 15, hh = lane >> 4;

  for (int i = tid; i < 16 * HD; i += TPB) hA[i] = (_Float16)0.0f;
  for (int i = tid; i < 16 * 32; i += TPB) xbuf[i] = (_Float16)0.0f;  // K-pad stays 0

  _Float16* hcur = hA;
  _Float16* hnxt = hB;
  float kacc = 0.0f, nacc = 0.0f;
  __syncthreads();

#pragma unroll 1
  for (int t = 0; t < NSTEP; ++t) {
    // x[t+1] tile -> xbuf (rows 0..15, cols 0..1; rest is permanent zero pad)
    if (tid < 32) {
      const int row = tid & 15, c = tid >> 4;
      xbuf[row * 32 + c] =
          (_Float16)P.x[(size_t)(t + 1) * BB * XD + (size_t)(mBase + row) * XD + c];
    }
    __syncthreads();
    // phi_x layer 1: K=32(pad), N=256 -> t1
    for (int nt = wave; nt < 16; nt += WAVES) {
      v8f acc = {};
      acc = gemm_part(acc, xbuf, 32, P.w + OFF_PHIX1, 16, nt, 1, 0, lane);
      store_act(t1b, HD, nt * 16, acc, P.b_px1, true, lane);
    }
    __syncthreads();
    // phi_x layer 2 -> px
    for (int nt = wave; nt < 16; nt += WAVES) {
      v8f acc = {};
      acc = gemm_part(acc, t1b, HD, P.w + OFF_PHIX2, 16, nt, 8, 0, lane);
      store_act(pxb, HD, nt * 16, acc, P.b_px2, true, lane);
    }
    __syncthreads();
    // encoder layer 1: [px, h] K=512 -> t1   (no materialized concat)
    for (int nt = wave; nt < 16; nt += WAVES) {
      v8f acc = {};
      acc = gemm_part(acc, pxb, HD, P.w + OFF_ENC1, 16, nt, 8, 0, lane);
      acc = gemm_part(acc, hcur, HD, P.w + OFF_ENC1, 16, nt, 8, 8, lane);
      store_act(t1b, HD, nt * 16, acc, P.b_e1, true, lane);
    }
    __syncthreads();
    // encoder layer 2 -> t2
    for (int nt = wave; nt < 16; nt += WAVES) {
      v8f acc = {};
      acc = gemm_part(acc, t1b, HD, P.w + OFF_ENC2, 16, nt, 8, 0, lane);
      store_act(t2b, HD, nt * 16, acc, P.b_e2, true, lane);
    }
    __syncthreads();
    // enc mean / logvar -> f32 LDS (one 16-col tile per wave)
    {
      const int nt = wave;
      v8f am = {}, al = {};
      am = gemm_part(am, t2b, HD, P.w + OFF_ENCM, 4, nt, 8, 0, lane);
      al = gemm_part(al, t2b, HD, P.w + OFF_ENCL, 4, nt, 8, 0, lane);
      const float bm = P.b_em[nt * 16 + col], bl = P.b_el[nt * 16 + col];
#pragma unroll
      for (int r = 0; r < 8; ++r) {
        emb[(r + 8 * hh) * ZD + nt * 16 + col] = am[r] + bm;
        elb[(r + 8 * hh) * ZD + nt * 16 + col] = al[r] + bl;
      }
    }
    __syncthreads();
    // prior hidden -> t1
    for (int nt = wave; nt < 16; nt += WAVES) {
      v8f acc = {};
      acc = gemm_part(acc, hcur, HD, P.w + OFF_PR, 16, nt, 8, 0, lane);
      store_act(t1b, HD, nt * 16, acc, P.b_pr, true, lane);
    }
    __syncthreads();
    // prior mean/logvar + KLD accumulation (pm/pl never leave registers)
    {
      const int nt = wave;
      v8f am = {}, al = {};
      am = gemm_part(am, t1b, HD, P.w + OFF_PRM, 4, nt, 8, 0, lane);
      al = gemm_part(al, t1b, HD, P.w + OFF_PRL, 4, nt, 8, 0, lane);
      const float bm = P.b_pm[nt * 16 + col], bl = P.b_pl[nt * 16 + col];
#pragma unroll
      for (int r = 0; r < 8; ++r) {
        const int m = r + 8 * hh, n = nt * 16 + col;
        const float pmv = am[r] + bm, plv = al[r] + bl;
        const float emv = emb[m * ZD + n], elv = elb[m * ZD + n];
        const float d = emv - pmv;
        kacc += 0.5f * ((plv - elv) + __expf(elv - plv) + d * d * __expf(-plv));
      }
    }
    // reparameterized z -> f16 (reads emb/elb only; no conflict with kld stage)
    for (int i = tid; i < 16 * ZD; i += TPB) {
      const int row = i >> 6, cc = i & 63;
      const float zv =
          emb[i] + P.eps[(size_t)t * BB * ZD + (size_t)(mBase + row) * ZD + cc] *
                       __expf(0.5f * elb[i]);
      zbuf[i] = (_Float16)zv;
    }
    __syncthreads();
    // phi_z -> pz
    for (int nt = wave; nt < 16; nt += WAVES) {
      v8f acc = {};
      acc = gemm_part(acc, zbuf, ZD, P.w + OFF_PHIZ, 16, nt, 2, 0, lane);
      store_act(pzb, HD, nt * 16, acc, P.b_pz, true, lane);
    }
    __syncthreads();
    // decoder layer 1: [pz, h] -> t1
    for (int nt = wave; nt < 16; nt += WAVES) {
      v8f acc = {};
      acc = gemm_part(acc, pzb, HD, P.w + OFF_DEC1, 16, nt, 8, 0, lane);
      acc = gemm_part(acc, hcur, HD, P.w + OFF_DEC1, 16, nt, 8, 8, lane);
      store_act(t1b, HD, nt * 16, acc, P.b_d1, true, lane);
    }
    __syncthreads();
    // decoder layer 2 -> t2
    for (int nt = wave; nt < 16; nt += WAVES) {
      v8f acc = {};
      acc = gemm_part(acc, t1b, HD, P.w + OFF_DEC2, 16, nt, 8, 0, lane);
      store_act(t2b, HD, nt * 16, acc, P.b_d2, true, lane);
    }
    __syncthreads();
    // dec mean/logvar (N=2 in one padded tile) + NLL  [wave 0 only; uniform branch]
    if (wave == 0) {
      v8f am = {}, al = {};
      am = gemm_part(am, t2b, HD, P.w + OFF_DECM, 1, 0, 8, 0, lane);
      al = gemm_part(al, t2b, HD, P.w + OFF_DECL, 1, 0, 8, 0, lane);
      if (col < XD) {
        const float bm = P.b_dm[col], bl = P.b_dl[col];
#pragma unroll
        for (int r = 0; r < 8; ++r) {
          const int m = r + 8 * hh;
          const float dm = fminf(10.0f, fmaxf(-10.0f, am[r] + bm));  // Hardtanh
          const float dl = al[r] + bl;
          const float xv =
              P.x[(size_t)(t + 1) * BB * XD + (size_t)(mBase + m) * XD + col];
          const float d = xv - dm;
          nacc += 0.5f * (d * d * __expf(-dl) + dl);
        }
      }
    }
    // GRU cell (reads pxb/pzb/hcur, writes hnxt; overlaps with dec stats)
#pragma unroll 1
    for (int j = wave; j < 16; j += WAVES) {
      v8f ir = {}, iz = {}, in_ = {}, hr = {}, hz = {}, hn = {};
      ir  = gemm_part(ir,  pxb, HD, P.w + OFF_GIH, 48, j,      8, 0, lane);
      ir  = gemm_part(ir,  pzb, HD, P.w + OFF_GIH, 48, j,      8, 8, lane);
      iz  = gemm_part(iz,  pxb, HD, P.w + OFF_GIH, 48, 16 + j, 8, 0, lane);
      iz  = gemm_part(iz,  pzb, HD, P.w + OFF_GIH, 48, 16 + j, 8, 8, lane);
      in_ = gemm_part(in_, pxb, HD, P.w + OFF_GIH, 48, 32 + j, 8, 0, lane);
      in_ = gemm_part(in_, pzb, HD, P.w + OFF_GIH, 48, 32 + j, 8, 8, lane);
      hr  = gemm_part(hr,  hcur, HD, P.w + OFF_GHH, 48, j,      8, 0, lane);
      hz  = gemm_part(hz,  hcur, HD, P.w + OFF_GHH, 48, 16 + j, 8, 0, lane);
      hn  = gemm_part(hn,  hcur, HD, P.w + OFF_GHH, 48, 32 + j, 8, 0, lane);
#pragma unroll
      for (int r = 0; r < 8; ++r) {
        const int m = r + 8 * hh, n = j * 16 + col;
        const float rg = 1.0f / (1.0f + __expf(-(ir[r] + hr[r])));
        const float zg = 1.0f / (1.0f + __expf(-(iz[r] + hz[r])));
        const float ng = tanhf(in_[r] + rg * hn[r]);
        const float ho = (float)hcur[m * HD + n];
        hnxt[m * HD + n] = (_Float16)((1.0f - zg) * ng + zg * ho);
      }
    }
    __syncthreads();  // hnxt complete; all hcur readers done
    _Float16* tsw = hcur; hcur = hnxt; hnxt = tsw;  // double-buffered h
  }

  // final hidden state -> d_out[2:]
  for (int i = tid; i < 16 * HD; i += TPB) {
    const int row = i >> 8, cc = i & 255;
    P.out[2 + (size_t)(mBase + row) * HD + cc] = (float)hcur[i];
  }
  atomicAdd(&P.out[0], kacc * (1.0f / BB));
  atomicAdd(&P.out[1], nacc * (1.0f / BB));
  if (blockIdx.x == 0 && tid == 0) {
    atomicAdd(&P.out[0], (float)NSTEP * (-0.5f * (float)ZD));       // -ZD/2 per step
    atomicAdd(&P.out[1], (float)NSTEP * 0.5f * (float)XD * LOG2PI); // const NLL term
  }
}

extern "C" void kernel_launch(void* const* d_in, const int* in_sizes, int n_in,
                              void* d_out, int out_size, void* d_ws, size_t ws_size,
                              hipStream_t stream) {
  (void)in_sizes; (void)n_in; (void)out_size; (void)ws_size;
  _Float16* w = (_Float16*)d_ws;

  auto pack = [&](int inIdx, size_t off, int N, int K, int nT, int kS) {
    const int total  = kS * nT * 512;
    const int blocks = (total + 255) / 256;
    vrnn_pack_w<<<blocks, 256, 0, stream>>>((const float*)d_in[inIdx], w + off,
                                            N, K, nT, kS);
  };
  pack(2,  OFF_PHIX1, 256, 2,   16, 1);
  pack(4,  OFF_PHIX2, 256, 256, 16, 8);
  pack(6,  OFF_PHIZ,  256, 64,  16, 2);
  pack(8,  OFF_ENC1,  256, 512, 16, 16);
  pack(10, OFF_ENC2,  256, 256, 16, 8);
  pack(12, OFF_ENCM,  64,  256, 4,  8);
  pack(14, OFF_ENCL,  64,  256, 4,  8);
  pack(16, OFF_PR,    256, 256, 16, 8);
  pack(18, OFF_PRM,   64,  256, 4,  8);
  pack(20, OFF_PRL,   64,  256, 4,  8);
  pack(22, OFF_DEC1,  256, 512, 16, 16);
  pack(24, OFF_DEC2,  256, 256, 16, 8);
  pack(26, OFF_DECM,  2,   256, 1,  8);
  pack(28, OFF_DECL,  2,   256, 1,  8);
  pack(30, OFF_GIH,   768, 512, 48, 16);
  pack(31, OFF_GHH,   768, 256, 48, 8);

  vrnn_init_out<<<1, 32, 0, stream>>>((float*)d_out);

  VrnnArgs A;
  A.x = (const float*)d_in[0];
  A.eps = (const float*)d_in[1];
  A.b_px1 = (const float*)d_in[3];
  A.b_px2 = (const float*)d_in[5];
  A.b_pz  = (const float*)d_in[7];
  A.b_e1  = (const float*)d_in[9];
  A.b_e2  = (const float*)d_in[11];
  A.b_em  = (const float*)d_in[13];
  A.b_el  = (const float*)d_in[15];
  A.b_pr  = (const float*)d_in[17];
  A.b_pm  = (const float*)d_in[19];
  A.b_pl  = (const float*)d_in[21];
  A.b_d1  = (const float*)d_in[23];
  A.b_d2  = (const float*)d_in[25];
  A.b_dm  = (const float*)d_in[27];
  A.b_dl  = (const float*)d_in[29];
  A.w = w;
  A.out = (float*)d_out;

  vrnn_main<<<BB / 16, TPB, 0, stream>>>(A);
}
